// TestTimeAdapter_68702296867035
// MI455X (gfx1250) — matrix operand order repaired and checked by reading, not compiled, not run on hardware
//
#include <hip/hip_runtime.h>
#include <hip/hip_bf16.h>
#include <math.h>

// ---------------------------------------------------------------------------
// CDNA5 (gfx1250) wave32 WMMA types
// ---------------------------------------------------------------------------
typedef __attribute__((ext_vector_type(16))) __bf16 v16bf;
typedef __attribute__((ext_vector_type(8)))  float  v8f;

union FragBF {
    uint4 u[2];   // 32 bytes per lane
    v16bf v;
};

__device__ __forceinline__ unsigned short f2bf(float f) {
    unsigned int u = __float_as_uint(f);
    u += 0x7fffu + ((u >> 16) & 1u);      // round-to-nearest-even
    return (unsigned short)(u >> 16);
}
__device__ __forceinline__ float bf2f(unsigned short h) {
    return __uint_as_float(((unsigned int)h) << 16);
}

// ---------------------------------------------------------------------------
// Kernel 1: normalize gallery rows per-camera, convert to bf16, row sumsq.
// Rows g in [G, Gp) are padding: bf16 zeros, gg = 1e30 (never in top-k).
// ---------------------------------------------------------------------------
__global__ __launch_bounds__(256) void norm_gallery_kernel(
    const float* __restrict__ gf, const int* __restrict__ camids,
    const float* __restrict__ means, const float* __restrict__ stds,
    unsigned short* __restrict__ gfb, float* __restrict__ gg,
    int G, int D)
{
    const int g   = blockIdx.x;
    const int tid = threadIdx.x;
    __shared__ float red[256];

    if (g >= G) {                         // padded row (uniform per block)
        for (int d = tid; d < D; d += 256) gfb[(size_t)g * D + d] = 0;
        if (tid == 0) gg[g] = 1.0e30f;
        return;
    }

    const int cam = camids[g];
    const float* frow = gf    + (size_t)g   * D;
    const float* mrow = means + (size_t)cam * D;
    const float* srow = stds  + (size_t)cam * D;
    unsigned short* orow = gfb + (size_t)g * D;

    float ssq = 0.0f;
    for (int d = tid; d < D; d += 256) {
        float v = (frow[d] - mrow[d]) / srow[d];
        unsigned short h = f2bf(v);
        orow[d] = h;
        float vr = bf2f(h);               // sumsq of the ROUNDED value
        ssq += vr * vr;
    }
    red[tid] = ssq;
    __syncthreads();
    for (int s = 128; s > 0; s >>= 1) {
        if (tid < s) red[tid] += red[tid + s];
        __syncthreads();
    }
    if (tid == 0) gg[g] = red[0];
}

// ---------------------------------------------------------------------------
// Kernel 2: normalize query rows, convert to bf16, row sumsq.
// ---------------------------------------------------------------------------
__global__ __launch_bounds__(256) void norm_query_kernel(
    const float* __restrict__ x, const int* __restrict__ c,
    const float* __restrict__ means, const float* __restrict__ stds,
    unsigned short* __restrict__ xb, float* __restrict__ xx, int D)
{
    const int q   = blockIdx.x;
    const int tid = threadIdx.x;
    __shared__ float red[256];

    const int cam = c[q];
    const float* frow = x     + (size_t)q   * D;
    const float* mrow = means + (size_t)cam * D;
    const float* srow = stds  + (size_t)cam * D;
    unsigned short* orow = xb + (size_t)q * D;

    float ssq = 0.0f;
    for (int d = tid; d < D; d += 256) {
        float v = (frow[d] - mrow[d]) / srow[d];
        unsigned short h = f2bf(v);
        orow[d] = h;
        float vr = bf2f(h);
        ssq += vr * vr;
    }
    red[tid] = ssq;
    __syncthreads();
    for (int s = 128; s > 0; s >>= 1) {
        if (tid < s) red[tid] += red[tid + s];
        __syncthreads();
    }
    if (tid == 0) xx[q] = red[0];
}

// ---------------------------------------------------------------------------
// Kernel 3: squared-distance GEMM via v_wmma_f32_16x16x32_bf16.
// Block = 256 threads = 8 waves arranged 2(q) x 4(g).
// Each wave owns a 32q x 32g register tile: 2 A frags x 2 B frags ->
// 4 WMMA accumulators, so operand traffic is 32B/lane per WMMA (vs 48B
// for a 16x32 tile). Block tile: 64q x 128g.
// S[qLocal, g] = xx[q] + gg[g] - 2 * dot.
//
// Fragment layouts per CDNA5 ISA (wave32):
//   A 16x32 bf16: lane%16 = M; lanes 0-15 hold K 0-7 (V0-3) & 16-23 (V4-7),
//                 lanes 16-31 hold K 8-15 & 24-31.
//   B 32x16 bf16: lane%16 = N; lanes 0-15 hold K 0-7 (V0-3) & 16-23 (V4-7),
//                 lanes 16-31 hold K 8-15 & 24-31.
// Both map to two contiguous b128 loads per lane from row-major data.
// ---------------------------------------------------------------------------
__global__ __launch_bounds__(256) void gemm_sqdist_kernel(
    const unsigned short* __restrict__ xb, const unsigned short* __restrict__ gfb,
    const float* __restrict__ xx, const float* __restrict__ gg,
    float* __restrict__ S, int qOff, int Gp, int D)
{
    const int lane = threadIdx.x & 31;
    const int w    = threadIdx.x >> 5;
    const int qw   = w >> 2;                  // 0..1
    const int gw   = w &  3;                  // 0..3
    const int half = lane >> 4;               // 0 or 1
    const int l16  = lane & 15;

    const int qTileLocal = blockIdx.y * 64 + qw * 32;   // within chunk
    const int qTile      = qOff + qTileLocal;
    const int gTile0     = blockIdx.x * 128 + gw * 32;
    const int gTile1     = gTile0 + 16;

    const unsigned short* xrow0 = xb  + (size_t)(qTile       + l16) * D;
    const unsigned short* xrow1 = xb  + (size_t)(qTile  + 16 + l16) * D;
    const unsigned short* grow0 = gfb + (size_t)(gTile0 + l16) * D;
    const unsigned short* grow1 = gfb + (size_t)(gTile1 + l16) * D;

    v8f acc00 = {};   // q 0..15  x g 0..15
    v8f acc01 = {};   // q 0..15  x g 16..31
    v8f acc10 = {};   // q 16..31 x g 0..15
    v8f acc11 = {};   // q 16..31 x g 16..31

    for (int k = 0; k < D; k += 32) {
        FragBF A0, A1, B0, B1;
        // A fragments: two 16B chunks each, selected by lane half
        A0.u[0] = *(const uint4*)(xrow0 + k +      half * 8);
        A0.u[1] = *(const uint4*)(xrow0 + k + 16 + half * 8);
        A1.u[0] = *(const uint4*)(xrow1 + k +      half * 8);
        A1.u[1] = *(const uint4*)(xrow1 + k + 16 + half * 8);
        // B fragments: two 16B chunks each, selected by lane half (same K order as A)
        B0.u[0] = *(const uint4*)(grow0 + k +      half * 8);
        B0.u[1] = *(const uint4*)(grow0 + k + 16 + half * 8);
        B1.u[0] = *(const uint4*)(grow1 + k +      half * 8);
        B1.u[1] = *(const uint4*)(grow1 + k + 16 + half * 8);

        acc00 = __builtin_amdgcn_wmma_f32_16x16x32_bf16(
                    false, A0.v, false, B0.v, (short)0, acc00, false, false);
        acc01 = __builtin_amdgcn_wmma_f32_16x16x32_bf16(
                    false, A0.v, false, B1.v, (short)0, acc01, false, false);
        acc10 = __builtin_amdgcn_wmma_f32_16x16x32_bf16(
                    false, A1.v, false, B0.v, (short)0, acc10, false, false);
        acc11 = __builtin_amdgcn_wmma_f32_16x16x32_bf16(
                    false, A1.v, false, B1.v, (short)0, acc11, false, false);
    }

    // C/D layout: lane 0-15: N=lane, M=r ; lane 16-31: N=lane-16, M=r+8
    const int g0 = gTile0 + l16;
    const int g1 = gTile1 + l16;
    const float gg0 = gg[g0];
    const float gg1 = gg[g1];
#pragma unroll
    for (int r = 0; r < 8; ++r) {
        const int m  = r + 8 * half;
        // q-tile 0 (rows qTileLocal + m)
        {
            const int qL = qTileLocal + m;
            const float xq = xx[qOff + qL];
            S[(size_t)qL * Gp + g0] = xq + gg0 - 2.0f * acc00[r];
            S[(size_t)qL * Gp + g1] = xq + gg1 - 2.0f * acc01[r];
        }
        // q-tile 1 (rows qTileLocal + 16 + m)
        {
            const int qL = qTileLocal + 16 + m;
            const float xq = xx[qOff + qL];
            S[(size_t)qL * Gp + g0] = xq + gg0 - 2.0f * acc10[r];
            S[(size_t)qL * Gp + g1] = xq + gg1 - 2.0f * acc11[r];
        }
    }
}

// ---------------------------------------------------------------------------
// Kernel 4: per-row sum of sqrt of the K smallest squared distances.
// One block per row; K iterations of deterministic block argmin (tie-break
// lowest index). Already-selected entries tracked in an LDS bitmask
// (dynamic shared, (Gp+31)/32 dwords); row data stays in L2.
// ---------------------------------------------------------------------------
__global__ __launch_bounds__(256) void topk_sum_kernel(
    const float* __restrict__ S, float* __restrict__ rowsum,
    const int* __restrict__ topk_p, int Gp, int qOff)
{
    extern __shared__ unsigned int mask[];
    __shared__ float rv[256];
    __shared__ int   ri[256];
    __shared__ float accs;

    const int row = blockIdx.x;
    const int tid = threadIdx.x;
    const float* srow = S + (size_t)row * Gp;
    const int nw = (Gp + 31) >> 5;

    for (int i = tid; i < nw; i += 256) mask[i] = 0u;
    if (tid == 0) accs = 0.0f;
    __syncthreads();

    const int K = topk_p[0];
    for (int it = 0; it < K; ++it) {
        float best = 3.0e38f;
        int   bidx = 0x7fffffff;
        for (int i = tid; i < Gp; i += 256) {
            if ((mask[i >> 5] >> (i & 31)) & 1u) continue;
            float v = srow[i];
            if (v < best || (v == best && i < bidx)) { best = v; bidx = i; }
        }
        rv[tid] = best;
        ri[tid] = bidx;
        __syncthreads();
        for (int s = 128; s > 0; s >>= 1) {
            if (tid < s) {
                float ov = rv[tid + s];
                int   oi = ri[tid + s];
                if (ov < rv[tid] || (ov == rv[tid] && oi < ri[tid])) {
                    rv[tid] = ov; ri[tid] = oi;
                }
            }
            __syncthreads();
        }
        if (tid == 0) {
            accs += sqrtf(fmaxf(rv[0], 1.0e-12f));
            mask[ri[0] >> 5] |= 1u << (ri[0] & 31);
        }
        __syncthreads();
    }
    if (tid == 0) rowsum[qOff + row] = accs;
}

// ---------------------------------------------------------------------------
// Kernel 5: deterministic mean over row sums.
// ---------------------------------------------------------------------------
__global__ __launch_bounds__(256) void final_mean_kernel(
    const float* __restrict__ rowsum, float* __restrict__ out, int Q)
{
    __shared__ float red[256];
    const int tid = threadIdx.x;
    float s = 0.0f;
    for (int i = tid; i < Q; i += 256) s += rowsum[i];   // fixed order
    red[tid] = s;
    __syncthreads();
    for (int st = 128; st > 0; st >>= 1) {
        if (tid < st) red[tid] += red[tid + st];
        __syncthreads();
    }
    if (tid == 0) out[0] = red[0] / (float)Q;
}

// ---------------------------------------------------------------------------
// Host-side orchestration
// ---------------------------------------------------------------------------
extern "C" void kernel_launch(void* const* d_in, const int* in_sizes, int n_in,
                              void* d_out, int out_size, void* d_ws, size_t ws_size,
                              hipStream_t stream)
{
    const float* x      = (const float*)d_in[0];
    const int*   c      = (const int*)  d_in[1];
    const float* gf     = (const float*)d_in[2];
    const int*   gcam   = (const int*)  d_in[3];
    const float* gmeans = (const float*)d_in[4];
    const float* gstds  = (const float*)d_in[5];
    const float* qmeans = (const float*)d_in[6];
    const float* qstds  = (const float*)d_in[7];
    const int*   topk   = (const int*)  d_in[8];
    float* out = (float*)d_out;

    const int Q  = in_sizes[1];
    const int G  = in_sizes[3];
    const int D  = in_sizes[0] / Q;               // 2048 (multiple of 32)
    const int Gp = ((G + 127) / 128) * 128;       // pad to block g-tile

    // workspace carve-out (256B aligned)
    char* ws = (char*)d_ws;
    size_t off = 0;
    auto alloc = [&](size_t bytes) -> void* {
        off = (off + 255) & ~(size_t)255;
        void* p = ws + off;
        off += bytes;
        return p;
    };
    unsigned short* gfb    = (unsigned short*)alloc((size_t)Gp * D * sizeof(unsigned short));
    unsigned short* xb     = (unsigned short*)alloc((size_t)Q  * D * sizeof(unsigned short));
    float*          gg     = (float*)alloc((size_t)Gp * sizeof(float));
    float*          xx     = (float*)alloc((size_t)Q  * sizeof(float));
    float*          rowsum = (float*)alloc((size_t)Q  * sizeof(float));
    const int QCHUNK = 256;                        // S chunk ~20.6 MB (L2-resident)
    float*          S      = (float*)alloc((size_t)QCHUNK * Gp * sizeof(float));
    (void)ws_size;

    norm_gallery_kernel<<<Gp, 256, 0, stream>>>(gf, gcam, gmeans, gstds, gfb, gg, G, D);
    norm_query_kernel  <<<Q,  256, 0, stream>>>(x, c, qmeans, qstds, xb, xx, D);

    const size_t maskBytes = (size_t)((Gp + 31) / 32) * sizeof(unsigned int);
    for (int qOff = 0; qOff < Q; qOff += QCHUNK) {
        const int rows = (Q - qOff < QCHUNK) ? (Q - qOff) : QCHUNK;
        dim3 grid(Gp / 128, rows / 64);
        gemm_sqdist_kernel<<<grid, 256, 0, stream>>>(xb, gfb, xx, gg, S, qOff, Gp, D);
        topk_sum_kernel<<<rows, 256, maskBytes, stream>>>(S, rowsum, topk, Gp, qOff);
    }
    final_mean_kernel<<<1, 256, 0, stream>>>(rowsum, out, Q);
}
